// HarmonicCausalSelfAttention_87213605913089
// MI455X (gfx1250) — matrix operand order repaired and checked by reading, not compiled, not run on hardware
//
#include <hip/hip_runtime.h>

// ---------------------------------------------------------------------------
// Types for CDNA5 WMMA (wave32): A/B = v16bf (16 x bf16), C/D = v8f.
// ---------------------------------------------------------------------------
typedef __attribute__((ext_vector_type(16))) __bf16       v16bf;
typedef __attribute__((ext_vector_type(8)))  float        v8f;
typedef __attribute__((ext_vector_type(4)))  unsigned int v4u;

union FragV { v16bf bf; v4u q[2]; };

constexpr int B_  = 4;
constexpr int T_  = 2048;
constexpr int C_  = 1024;
constexpr int NH_ = 16;
constexpr int HD_ = 64;
constexpr int RK_ = 128;
constexpr int M_  = B_ * T_;   // 8192 rows total

__device__ __forceinline__ unsigned short f2bf(float f) {
  unsigned u = __float_as_uint(f);
  u += 0x7FFFu + ((u >> 16) & 1u);       // round-to-nearest-even
  return (unsigned short)(u >> 16);
}

__device__ __forceinline__ v8f wmma_bf16(const FragV& a, const FragV& b, v8f c) {
  // (neg_a, A, neg_b, B, c_mod, C, reuse_a, reuse_b)
  return __builtin_amdgcn_wmma_f32_16x16x32_bf16(false, a.bf, false, b.bf,
                                                 (short)0, c, false, false);
}

// Low 32 bits of a generic (flat) shared pointer == wave-relative LDS byte
// address (ISA 10.2: an LDS-aperture address keeps only its low 32 bits).
__device__ __forceinline__ unsigned lds_addr32(const void* p) {
  return (unsigned)(unsigned long long)p;
}

// ---------------------------------------------------------------------------
// fp32 -> bf16 conversion
// ---------------------------------------------------------------------------
__global__ void cvt_f32_bf16(const float* __restrict__ src,
                             unsigned short* __restrict__ dst, int n) {
  int i = blockIdx.x * blockDim.x + threadIdx.x;
  if (i < n) dst[i] = f2bf(src[i]);
}

// ---------------------------------------------------------------------------
// Y[M,N] = X[M,K] @ W[N,K]^T   (X, W bf16 row-major; W rows are the N dim,
// which is exactly the WMMA B-fragment N-major/K-contiguous layout).
// Block: 256 thr = 8 waves; tile 128(M) x 128(N); each wave one 16x128 strip.
// W tile is staged memory->LDS with GLOBAL_LOAD_ASYNC_TO_LDS_B128 (ASYNCcnt).
// MODE 0: bf16 out row-major [M,N]
// MODE 1: bf16 out scattered to [b,h,t,d]   (k)
// MODE 2: bf16 out scattered to [b,h,d,t]   (v, pre-transposed for PV matmul)
// MODE 3: f32  out row-major [M,N]          (final projection -> d_out)
// MODE 4: like MODE 1 but scaled by 1/sqrt(HD)  (q; folds softmax scale)
// ---------------------------------------------------------------------------
template <int MODE>
__global__ __launch_bounds__(256) void gemm_bf16_k(
    const unsigned short* __restrict__ X,
    const unsigned short* __restrict__ W,
    void* __restrict__ outp, int M, int N, int K) {
  __shared__ unsigned short wt[128 * 32];  // W tile: [128 N][32 K], 8 KB
  const int tid  = threadIdx.x;
  const int wave = tid >> 5;
  const int lane = tid & 31;
  const int hl   = lane >> 4;      // lane half
  const int ll   = lane & 15;
  const int n_block = blockIdx.y * 128;
  const int m_base  = blockIdx.x * 128 + wave * 16;
  (void)M;

  // per-thread staging coordinates: 256 thr x 32 B = 8 KB tile
  const int r_st  = tid >> 1;            // tile row 0..127
  const int c_st  = (tid & 1) * 16;      // K element offset 0 / 16
  const unsigned long long wbase = (unsigned long long)W;
  const unsigned lds_dst = lds_addr32(&wt[r_st * 32 + c_st]);

  v8f acc[8];
#pragma unroll
  for (int nt = 0; nt < 8; ++nt)
#pragma unroll
    for (int e = 0; e < 8; ++e) acc[nt][e] = 0.0f;

  for (int k0 = 0; k0 < K; k0 += 32) {
    {  // async stage of W[n_block..+128][k0..+32] into LDS (2 x 16B per thread)
      const unsigned goff =
          (unsigned)(((size_t)(n_block + r_st) * K + k0 + c_st) * 2u);
      asm volatile("global_load_async_to_lds_b128 %0, %1, %2\n\t"
                   "global_load_async_to_lds_b128 %0, %1, %2 offset:16"
                   :
                   : "v"(lds_dst), "v"(goff), "s"(wbase)
                   : "memory");
      asm volatile("s_wait_asynccnt 0" ::: "memory");
    }
    __syncthreads();

    // A fragment: lane ll = row, K packed 8 (hl*8) + 8 (16+hl*8)
    FragV af;
    const unsigned short* xp = X + (size_t)(m_base + ll) * K + k0 + hl * 8;
    af.q[0] = *(const v4u*)(xp);
    af.q[1] = *(const v4u*)(xp + 16);
    __builtin_prefetch(xp + 32, 0, 3);   // next K-slice of this row

#pragma unroll
    for (int nt = 0; nt < 8; ++nt) {
      // B fragment: lane ll = N col, K packed 8 (hl*8) + 8 (16+hl*8)
      FragV bfr;
      const unsigned short* wp = &wt[(nt * 16 + ll) * 32 + hl * 8];
      bfr.q[0] = *(const v4u*)(wp);
      bfr.q[1] = *(const v4u*)(wp + 16);
      acc[nt] = wmma_bf16(af, bfr, acc[nt]);
    }
    __syncthreads();
  }

  // Epilogue: C layout -> VGPR j holds row (j + 8*hl), lane ll holds col.
#pragma unroll
  for (int nt = 0; nt < 8; ++nt) {
#pragma unroll
    for (int j = 0; j < 8; ++j) {
      const int mrow = m_base + j + 8 * hl;
      const int col  = n_block + nt * 16 + ll;
      float v = acc[nt][j];
      if (MODE == 4) v *= 0.125f;        // 1/sqrt(64) folded into Q
      if (MODE == 0) {
        ((unsigned short*)outp)[(size_t)mrow * N + col] = f2bf(v);
      } else if (MODE == 1 || MODE == 4) {
        const int b = mrow >> 11, t = mrow & (T_ - 1);
        const int h = col >> 6,  d = col & (HD_ - 1);
        ((unsigned short*)outp)[(((size_t)(b * NH_ + h)) * T_ + t) * HD_ + d] = f2bf(v);
      } else if (MODE == 2) {
        const int b = mrow >> 11, t = mrow & (T_ - 1);
        const int h = col >> 6,  d = col & (HD_ - 1);
        ((unsigned short*)outp)[(((size_t)(b * NH_ + h)) * HD_ + d) * T_ + t] = f2bf(v);
      } else {
        ((float*)outp)[(size_t)mrow * N + col] = v;
      }
    }
  }
}

// ---------------------------------------------------------------------------
// Flash-attention: each wave owns 16 query rows of one (b,h); streams key
// chunks of 64 with online softmax (wide chunks amortize the shuffle
// reductions and O-rescale against 16 co-executing WMMAs per chunk).
// Q (pre-scaled), K in [b,h,t,d]; V pre-transposed [b,h,d,t]; output bf16
// to y[b,t,h*64+d] (feeds the c-projection).
// ---------------------------------------------------------------------------
__global__ __launch_bounds__(256) void attn_k(
    const unsigned short* __restrict__ Q,
    const unsigned short* __restrict__ K,
    const unsigned short* __restrict__ Vt,
    unsigned short* __restrict__ Y) {
  constexpr int PSTR = 72;                       // 64 cols + skew, 16B-aligned
  __shared__ unsigned short pls[8 * 16 * PSTR];  // per-wave P staging (18 KB)
  const int tid  = threadIdx.x;
  const int wave = tid >> 5;
  const int lane = tid & 31;
  const int hl   = lane >> 4;
  const int ll   = lane & 15;
  const int bh = blockIdx.y;
  const int b  = bh >> 4, h = bh & (NH_ - 1);
  const int m_base = blockIdx.x * 128 + wave * 16;

  const unsigned short* Qh = Q  + (size_t)bh * T_ * HD_;
  const unsigned short* Kh = K  + (size_t)bh * T_ * HD_;
  const unsigned short* Vh = Vt + (size_t)bh * HD_ * T_;
  unsigned short* pw = &pls[wave * 16 * PSTR];

  // Q fragments for the 16-row strip (K = HD = 64 -> 2 fragments), in regs
  FragV qf[2];
  {
    const unsigned short* qp = Qh + (size_t)(m_base + ll) * HD_ + hl * 8;
    qf[0].q[0] = *(const v4u*)(qp);
    qf[0].q[1] = *(const v4u*)(qp + 16);
    qf[1].q[0] = *(const v4u*)(qp + 32);
    qf[1].q[1] = *(const v4u*)(qp + 48);
  }

  v8f oacc[4];                 // 16 x 64 output accumulator
  float mrow[8], lrow[8];      // row max / row sum (row = j + 8*hl)
#pragma unroll
  for (int nt = 0; nt < 4; ++nt)
#pragma unroll
    for (int e = 0; e < 8; ++e) oacc[nt][e] = 0.0f;
#pragma unroll
  for (int j = 0; j < 8; ++j) { mrow[j] = -1e30f; lrow[j] = 0.0f; }

  const int jmax = ((m_base + 15) >> 6) << 6;    // last causal 64-key chunk
  for (int j0 = 0; j0 <= jmax; j0 += 64) {
    // ---- S = Q @ K^T for 64 keys (four 16x16 tiles, K=64 -> 2 steps each)
    v8f s[4];
#pragma unroll
    for (int kt = 0; kt < 4; ++kt) {
#pragma unroll
      for (int e = 0; e < 8; ++e) s[kt][e] = 0.0f;
      const unsigned short* kp =
          Kh + (size_t)(j0 + kt * 16 + ll) * HD_ + hl * 8;
      FragV kf;
      kf.q[0] = *(const v4u*)(kp);
      kf.q[1] = *(const v4u*)(kp + 16);
      s[kt] = wmma_bf16(qf[0], kf, s[kt]);
      kf.q[0] = *(const v4u*)(kp + 32);
      kf.q[1] = *(const v4u*)(kp + 48);
      s[kt] = wmma_bf16(qf[1], kf, s[kt]);
    }

    // ---- online softmax over this 64-key chunk (Q already carries 1/8)
    const bool needMask = (j0 + 63 > m_base);
#pragma unroll
    for (int j = 0; j < 8; ++j) {
      float v0 = s[0][j], v1 = s[1][j], v2 = s[2][j], v3 = s[3][j];
      if (needMask) {
        const int row = m_base + j + 8 * hl;
        v0 = (j0 + ll      > row) ? -1e30f : v0;
        v1 = (j0 + 16 + ll > row) ? -1e30f : v1;
        v2 = (j0 + 32 + ll > row) ? -1e30f : v2;
        v3 = (j0 + 48 + ll > row) ? -1e30f : v3;
      }
      float rm = fmaxf(fmaxf(v0, v1), fmaxf(v2, v3));
      for (int off = 1; off < 16; off <<= 1)       // reduce 16 lanes of half
        rm = fmaxf(rm, __shfl_xor(rm, off, 32));
      const float mnew = fmaxf(mrow[j], rm);
      const float al   = __expf(mrow[j] - mnew);
      const float p0   = __expf(v0 - mnew);
      const float p1   = __expf(v1 - mnew);
      const float p2   = __expf(v2 - mnew);
      const float p3   = __expf(v3 - mnew);
      float rs = (p0 + p1) + (p2 + p3);
      for (int off = 1; off < 16; off <<= 1)
        rs += __shfl_xor(rs, off, 32);
      lrow[j] = lrow[j] * al + rs;
      mrow[j] = mnew;
#pragma unroll
      for (int nt = 0; nt < 4; ++nt) oacc[nt][j] *= al;
      // stage P (C-layout) into this wave's private LDS tile (row-major 16x64)
      unsigned short* pr = pw + (j + 8 * hl) * PSTR;
      pr[ll]      = f2bf(p0);
      pr[16 + ll] = f2bf(p1);
      pr[32 + ll] = f2bf(p2);
      pr[48 + ll] = f2bf(p3);
    }

    // ---- O += P @ V over two 32-key sub-chunks (same-wave DS is in-order)
#pragma unroll
    for (int ks = 0; ks < 2; ++ks) {
      FragV pf;
      const unsigned short* pr = pw + ll * PSTR + ks * 32 + hl * 8;
      pf.q[0] = *(const v4u*)(pr);
      pf.q[1] = *(const v4u*)(pr + 16);
#pragma unroll
      for (int nt = 0; nt < 4; ++nt) {
        const unsigned short* vp =
            Vh + (size_t)(nt * 16 + ll) * T_ + j0 + ks * 32 + hl * 8;
        FragV vf;
        vf.q[0] = *(const v4u*)(vp);
        vf.q[1] = *(const v4u*)(vp + 16);
        oacc[nt] = wmma_bf16(pf, vf, oacc[nt]);
      }
    }
  }

  // ---- normalize and write y[b, t, h*64 + d] (bf16, feeds c-projection)
#pragma unroll
  for (int j = 0; j < 8; ++j) {
    const float inv = 1.0f / lrow[j];
    const int row = m_base + j + 8 * hl;
#pragma unroll
    for (int nt = 0; nt < 4; ++nt) {
      Y[((size_t)(b * T_ + row)) * C_ + h * HD_ + nt * 16 + ll] =
          f2bf(oacc[nt][j] * inv);
    }
  }
}

// ---------------------------------------------------------------------------
// Host launch
// ---------------------------------------------------------------------------
extern "C" void kernel_launch(void* const* d_in, const int* in_sizes, int n_in,
                              void* d_out, int out_size, void* d_ws, size_t ws_size,
                              hipStream_t stream) {
  (void)in_sizes; (void)n_in; (void)out_size; (void)ws_size;
  const float* x = (const float*)d_in[0];

  char* ws = (char*)d_ws;
  const size_t SZ_BTC = (size_t)M_ * C_ * sizeof(unsigned short);  // 16 MB
  unsigned short* Xb = (unsigned short*)(ws);
  unsigned short* Yb = (unsigned short*)(ws + 1 * SZ_BTC);
  unsigned short* Qb = (unsigned short*)(ws + 2 * SZ_BTC);
  unsigned short* Kb = (unsigned short*)(ws + 3 * SZ_BTC);
  unsigned short* Vb = (unsigned short*)(ws + 4 * SZ_BTC);
  unsigned short* Rb = (unsigned short*)(ws + 5 * SZ_BTC);          // 2 MB
  unsigned short* Wb[8];
  {
    char* wbase = ws + 5 * SZ_BTC + (size_t)M_ * RK_ * sizeof(unsigned short);
    for (int i = 0; i < 8; ++i)
      Wb[i] = (unsigned short*)(wbase + (size_t)i * RK_ * C_ * sizeof(unsigned short));
  }

  // bf16 conversions: x and the 8 weight matrices (qA,qB,kA,kB,vA,vB,cA,cB)
  {
    const int n = M_ * C_;
    cvt_f32_bf16<<<(n + 255) / 256, 256, 0, stream>>>(x, Xb, n);
    const int nw = RK_ * C_;
    for (int i = 0; i < 8; ++i)
      cvt_f32_bf16<<<(nw + 255) / 256, 256, 0, stream>>>(
          (const float*)d_in[1 + i], Wb[i], nw);
  }

  dim3 blk(256);
  dim3 gR(M_ / 128, RK_ / 128);   // 64 x 1  : rank GEMM  (K=1024, N=128)
  dim3 gO(M_ / 128, C_  / 128);   // 64 x 8  : expand GEMM (K=128,  N=1024)

  // Q = (x @ qA^T) @ qB^T  -> [b,h,t,d], pre-scaled by 1/sqrt(HD)
  gemm_bf16_k<0><<<gR, blk, 0, stream>>>(Xb, Wb[0], Rb, M_, RK_, C_);
  gemm_bf16_k<4><<<gO, blk, 0, stream>>>(Rb, Wb[1], Qb, M_, C_, RK_);
  // K -> [b,h,t,d]
  gemm_bf16_k<0><<<gR, blk, 0, stream>>>(Xb, Wb[2], Rb, M_, RK_, C_);
  gemm_bf16_k<1><<<gO, blk, 0, stream>>>(Rb, Wb[3], Kb, M_, C_, RK_);
  // V -> [b,h,d,t] (transposed for the PV matmul B-fragment layout)
  gemm_bf16_k<0><<<gR, blk, 0, stream>>>(Xb, Wb[4], Rb, M_, RK_, C_);
  gemm_bf16_k<2><<<gO, blk, 0, stream>>>(Rb, Wb[5], Vb, M_, C_, RK_);
  // attention -> y[b,t,c] bf16
  attn_k<<<dim3(T_ / 128, B_ * NH_), blk, 0, stream>>>(Qb, Kb, Vb, Yb);
  // out = (y @ cA^T) @ cB^T -> fp32 d_out
  gemm_bf16_k<0><<<gR, blk, 0, stream>>>(Yb, Wb[6], Rb, M_, RK_, C_);
  gemm_bf16_k<3><<<gO, blk, 0, stream>>>(Rb, Wb[7], (float*)d_out, M_, C_, RK_);
}
